// Elman_9698036155042
// MI455X (gfx1250) — matrix operand order repaired and checked by reading, not compiled, not run on hardware
//
#include <hip/hip_runtime.h>

#define BB 64
#define TT 512
#define EE 512
#define HH 512
#define OO 512
#define W1S (EE + HH)

typedef __bf16 bf16_t;
typedef bf16_t v16bf __attribute__((ext_vector_type(16)));
typedef float v8f __attribute__((ext_vector_type(8)));
typedef unsigned u32;
typedef u32 u32x4 __attribute__((ext_vector_type(4)));
typedef int i32x4 __attribute__((ext_vector_type(4)));
typedef int i32x8 __attribute__((ext_vector_type(8)));

static __device__ __forceinline__ v8f wmma_bf16(v16bf a, v16bf b, v8f c) {
  // D = A x B + C, f32 accumulate. (neg_a, A, neg_b, B, c_mod, C, reuse_a, reuse_b)
  return __builtin_amdgcn_wmma_f32_16x16x32_bf16(false, a, false, b, (short)0, c,
                                                 false, false);
}

static __device__ __forceinline__ v8f v8f_zero() {
  v8f v;
#pragma unroll
  for (int i = 0; i < 8; ++i) v[i] = 0.f;
  return v;
}

static __device__ __forceinline__ void split_bf16(float v, bf16_t& h, bf16_t& l) {
  h = (bf16_t)v;
  l = (bf16_t)(v - (float)h);
}

// A fragment (16x32 bf16, M=lane%16): element e -> K = k0 + e + lh*16
static __device__ __forceinline__ void load_a_frag(const float* __restrict__ row,
                                                   int k0, int lh, v16bf& hi, v16bf& lo) {
  const float* p0 = row + k0 + lh * 16;
#pragma unroll
  for (int i = 0; i < 8; ++i) {
    bf16_t h, l;
    split_bf16(p0[i], h, l);       hi[i] = h;     lo[i] = l;
    split_bf16(p0[8 + i], h, l);  hi[8 + i] = h; lo[8 + i] = l;
  }
}

// TDM: 1-D tile copy global -> LDS (nbytes must be a multiple of 4, <= 65535 elems).
// D# per CDNA5 ISA. 6-arg builtin form (clang-23 toolchain).
// Guarded so a missing builtin degrades gracefully (the caller also performs a
// plain copy afterwards for correctness).
static __device__ __forceinline__ void tdm_copy_to_lds(const void* gsrc, u32 lds_off,
                                                       u32 nbytes) {
#if __has_builtin(__builtin_amdgcn_tensor_load_to_lds) && \
    __has_builtin(__builtin_amdgcn_s_wait_tensorcnt)
  unsigned long long ga = (unsigned long long)(size_t)gsrc;
  u32 nelem = nbytes >> 2;  // 4-byte elements
  u32x4 g0;
  g0[0] = 1u;                                // count=1, user mode, no gather
  g0[1] = lds_off;                           // D#.lds_addr (bytes)
  g0[2] = (u32)ga;                           // global_addr[31:0]
  g0[3] = (u32)(ga >> 32) | (2u << 30);      // global_addr[56:32] | type=2 (image)
  i32x8 g1;
  g1[0] = (int)(2u << 16);                   // wg_mask=0, data_size=4B (code 2)
  g1[1] = (int)((nelem & 0xffffu) << 16);    // tensor_dim0[15:0] @bit48
  g1[2] = (int)((nelem >> 16) | (1u << 16)); // tensor_dim0[31:16] | tensor_dim1=1
  g1[3] = (int)((nelem & 0xffffu) << 16);    // tensor_dim1[31:16]=0 | tile_dim0
  g1[4] = 1;                                 // tile_dim1=1, tile_dim2=0
  g1[5] = (int)nelem;                        // tensor_dim0_stride[31:0]
  g1[6] = 0;                                 // stride msbs / dim1_stride lo
  g1[7] = 0;
  i32x4 z4;
  z4[0] = 0; z4[1] = 0; z4[2] = 0; z4[3] = 0;
  i32x8 z8;
#pragma unroll
  for (int i = 0; i < 8; ++i) z8[i] = 0;
  __builtin_amdgcn_tensor_load_to_lds(g0, g1, z4, z4, z8, 0);
  __builtin_amdgcn_s_wait_tensorcnt(0);
#else
  (void)gsrc; (void)lds_off; (void)nbytes;
#endif
}

// ---------------- Kernel 0: prepack W1/W2 into WMMA B-fragment order -----------------
// Fragment layout (per weight matrix, per hi/lo array), bf16 elements:
//   idx = ((ntile * KSTEPS + kstep) * 32 + lane) * 16 + e
//   where n = ntile*16 + lane%16, K = kstep*32 + (lane/16)*16 + e  (B[k][n] = W[n][k])
// Each lane's 16 elements are 32 contiguous bytes -> direct v16bf loads in the GEMMs.
#define NP1 (HH * W1S / 2)   // 262144 element-pairs for W1 (KSTEPS=32)
#define NP2 (OO * HH / 2)    // 131072 element-pairs for W2 (KSTEPS=16)
__global__ void __launch_bounds__(256) k_pack(const float* __restrict__ W1,
                                              const float* __restrict__ W2,
                                              bf16_t* __restrict__ w1hi,
                                              bf16_t* __restrict__ w1lo,
                                              bf16_t* __restrict__ w2hi,
                                              bf16_t* __restrict__ w2lo) {
  int p = (int)(blockIdx.x * 256u + threadIdx.x);
  if (p >= NP1 + NP2) return;
  const float* src;
  bf16_t *dhi, *dlo;
  if (p < NP1) {
    int fidx = p * 2;
    int e = fidx & 15, lane = (fidx >> 4) & 31, ks = (fidx >> 9) & 31, nt = fidx >> 14;
    int n = nt * 16 + (lane & 15);
    int K = ks * 32 + (lane >> 4) * 16 + e;
    src = W1 + (size_t)n * W1S + K;
    dhi = w1hi + fidx; dlo = w1lo + fidx;
  } else {
    int fidx = (p - NP1) * 2;
    int e = fidx & 15, lane = (fidx >> 4) & 31, ks = (fidx >> 9) & 15, nt = fidx >> 13;
    int n = nt * 16 + (lane & 15);
    int K = ks * 32 + (lane >> 4) * 16 + e;
    src = W2 + (size_t)n * HH + K;
    dhi = w2hi + fidx; dlo = w2lo + fidx;
  }
  float v0 = src[0], v1 = src[1];
  bf16_t h0, l0, h1, l1;
  split_bf16(v0, h0, l0);
  split_bf16(v1, h1, l1);
  dhi[0] = h0; dhi[1] = h1;
  dlo[0] = l0; dlo[1] = l1;
}

// ---------------- Kernel 1: Zx[t*B+b, n] = sum_k x[b,t,k]*W1[n,k] + b1[n] ------------
// rows = B*T = 32768 (r = t*64+b), cols = H = 512. Wave: 16 rows x 64 cols (4 col tiles).
__global__ void __launch_bounds__(128) k_zx(const float* __restrict__ x,
                                            const bf16_t* __restrict__ w1hi,
                                            const bf16_t* __restrict__ w1lo,
                                            const float* __restrict__ b1,
                                            float* __restrict__ zx) {
  int wave = (int)(blockIdx.x * 128u + threadIdx.x) >> 5;
  int lane = threadIdx.x & 31;
  int lh = lane >> 4, l16 = lane & 15;
  int rowtile = wave >> 3;   // 0..2047
  int colgrp  = wave & 7;    // 0..7
  int m_base = rowtile << 4;
  int n_base = colgrp << 6;
  int row = m_base + l16;            // GEMM row = t*64 + b
  int b = row & (BB - 1), t = row >> 6;
  const float* arow = x + ((size_t)b * TT + t) * EE;

  v8f acc[4];
#pragma unroll
  for (int ct = 0; ct < 4; ++ct) acc[ct] = v8f_zero();

  for (int k0 = 0; k0 < EE; k0 += 32) {
    int ks = k0 >> 5;
    v16bf ahi, alo;
    load_a_frag(arow, k0, lh, ahi, alo);
#pragma unroll
    for (int ct = 0; ct < 4; ++ct) {
      int nt = (n_base >> 4) + ct;
      size_t fb = ((size_t)(nt * 32 + ks) * 32 + lane) * 16;
      v16bf bhi = *(const v16bf*)(w1hi + fb);
      v16bf blo = *(const v16bf*)(w1lo + fb);
      acc[ct] = wmma_bf16(ahi, bhi, acc[ct]);
      acc[ct] = wmma_bf16(alo, bhi, acc[ct]);
      acc[ct] = wmma_bf16(ahi, blo, acc[ct]);
    }
  }
#pragma unroll
  for (int ct = 0; ct < 4; ++ct) {
    int n = n_base + ct * 16 + l16;
    float bias = b1[n];
#pragma unroll
    for (int r = 0; r < 8; ++r) {
      int m = m_base + r + lh * 8;
      zx[(size_t)m * HH + n] = acc[ct][r] + bias;
    }
  }
}

// ---------------- Kernel 2: persistent recurrence over T steps ----------------------
// 32 blocks (one 16-col tile each) x 4 waves (one 16-row tile each).
// W1h fragments (ksteps 16..31 of prepacked W1) staged into LDS: TDM async-tensor
// copy first (CDNA5 path), then a plain copy that guarantees correct contents.
// B fragments are direct v16bf LDS loads. Grid barrier (atomics) between timesteps.
__global__ void __launch_bounds__(128) k_rnn(const bf16_t* __restrict__ w1hi,
                                             const bf16_t* __restrict__ w1lo,
                                             const float* __restrict__ zx,
                                             float* __restrict__ hall,
                                             unsigned* __restrict__ syncv) {
  __shared__ __attribute__((aligned(32))) bf16_t ldshi[16 * 32 * 16];  // 16 KB
  __shared__ __attribute__((aligned(32))) bf16_t ldslo[16 * 32 * 16];  // 16 KB
  const int lane = threadIdx.x & 31;
  const int rt = threadIdx.x >> 5;         // row tile 0..3
  const int lh = lane >> 4, l16 = lane & 15;
  const int nt = (int)blockIdx.x;          // col tile
  const int n_base = nt << 4;

  // This col tile's W1h fragments are contiguous: bf16 offset nt*16384 + 8192 + [0,8192)
  const bf16_t* ghi_b = w1hi + (size_t)nt * 16384 + 8192;
  const bf16_t* glo_b = w1lo + (size_t)nt * 16384 + 8192;

  // --- TDM async-tensor staging (EXEC-independent; issue from wave 0 only) ---
  if (threadIdx.x == 0) {
    tdm_copy_to_lds(ghi_b, (u32)(size_t)&ldshi[0], 16 * 32 * 16 * 2);
    tdm_copy_to_lds(glo_b, (u32)(size_t)&ldslo[0], 16 * 32 * 16 * 2);
  }
  __syncthreads();
  // --- plain copy: guarantees correct LDS contents regardless of D# semantics ---
  {
    const unsigned* ghi = (const unsigned*)ghi_b;
    const unsigned* glo = (const unsigned*)glo_b;
    unsigned* lhp = (unsigned*)ldshi;
    unsigned* llp = (unsigned*)ldslo;
    for (int i = (int)threadIdx.x; i < 4096; i += 128) {
      lhp[i] = ghi[i];
      llp[i] = glo[i];
    }
  }
  __syncthreads();

  for (int t = 0; t < TT; ++t) {
    const float* hprev = hall + (size_t)t * (BB * HH);
    float* hcur = hall + (size_t)(t + 1) * (BB * HH);
    const float* zxt = zx + (size_t)t * (BB * HH);

    if (t + 1 < TT) {  // prefetch next step's Zx lines for this thread's outputs
      const float* nz = zx + (size_t)(t + 1) * (BB * HH);
      __builtin_prefetch(nz + (size_t)(rt * 16 + l16) * HH + n_base, 0, 1);
    }

    v8f acc = v8f_zero();
    for (int ks = 0; ks < 16; ++ks) {
      size_t lb = (size_t)(ks * 32 + lane) * 16;
      v16bf bhi = *(const v16bf*)(ldshi + lb);
      v16bf blo = *(const v16bf*)(ldslo + lb);
      const float* arow = hprev + (size_t)(rt * 16 + l16) * HH;
      v16bf ahi, alo;
      load_a_frag(arow, ks * 32, lh, ahi, alo);
      acc = wmma_bf16(ahi, bhi, acc);
      acc = wmma_bf16(alo, bhi, acc);
      acc = wmma_bf16(ahi, blo, acc);
    }
#pragma unroll
    for (int r = 0; r < 8; ++r) {
      int m = rt * 16 + r + lh * 8;
      int n = n_base + l16;
      size_t idx = (size_t)m * HH + n;
      float v = acc[r] + zxt[idx];
      hcur[idx] = v > 0.f ? v : 0.f;
    }

    // ---- grid barrier over gridDim.x blocks ----
    __syncthreads();
    if (threadIdx.x == 0) {
      __threadfence();
      unsigned gen = atomicAdd(&syncv[1], 0u);
      unsigned prev = atomicAdd(&syncv[0], 1u);
      if (prev == gridDim.x - 1) {
        atomicExch(&syncv[0], 0u);
        __threadfence();
        atomicAdd(&syncv[1], 1u);
      } else {
        while (atomicAdd(&syncv[1], 0u) == gen) __builtin_amdgcn_s_sleep(1);
      }
    }
    __syncthreads();
    __threadfence();  // acquire: invalidate near caches before reading new hprev
  }
}

// ---------------- Kernel 3: out[b,t,n] = sum_k H[t,b,k]*W2[n,k] + b2[n] -------------
__global__ void __launch_bounds__(128) k_out(const float* __restrict__ hall,
                                             const bf16_t* __restrict__ w2hi,
                                             const bf16_t* __restrict__ w2lo,
                                             const float* __restrict__ b2,
                                             float* __restrict__ out) {
  int wave = (int)(blockIdx.x * 128u + threadIdx.x) >> 5;
  int lane = threadIdx.x & 31;
  int lh = lane >> 4, l16 = lane & 15;
  int rowtile = wave >> 3;
  int colgrp  = wave & 7;
  int m_base = rowtile << 4;
  int n_base = colgrp << 6;
  // GEMM row m = t*64+b  ->  hall slice t+1, row b  ->  offset (m+64)*HH
  const float* arow = hall + (size_t)(m_base + l16 + BB) * HH;

  v8f acc[4];
#pragma unroll
  for (int ct = 0; ct < 4; ++ct) acc[ct] = v8f_zero();

  for (int k0 = 0; k0 < HH; k0 += 32) {
    int ks = k0 >> 5;
    v16bf ahi, alo;
    load_a_frag(arow, k0, lh, ahi, alo);
#pragma unroll
    for (int ct = 0; ct < 4; ++ct) {
      int nt = (n_base >> 4) + ct;
      size_t fb = ((size_t)(nt * 16 + ks) * 32 + lane) * 16;
      v16bf bhi = *(const v16bf*)(w2hi + fb);
      v16bf blo = *(const v16bf*)(w2lo + fb);
      acc[ct] = wmma_bf16(ahi, bhi, acc[ct]);
      acc[ct] = wmma_bf16(alo, bhi, acc[ct]);
      acc[ct] = wmma_bf16(ahi, blo, acc[ct]);
    }
  }
#pragma unroll
  for (int ct = 0; ct < 4; ++ct) {
    int n = n_base + ct * 16 + l16;
    float bias = b2[n];
#pragma unroll
    for (int r = 0; r < 8; ++r) {
      int m = m_base + r + lh * 8;
      int t = m >> 6, b = m & (BB - 1);
      out[((size_t)b * TT + t) * OO + n] = acc[ct][r] + bias;
    }
  }
}

// ---------------- helpers: init + h_final copy --------------------------------------
__global__ void k_init(float* __restrict__ hall, unsigned* __restrict__ syncv) {
  int i = (int)(blockIdx.x * 256u + threadIdx.x);
  if (i < BB * HH) hall[i] = 0.f;   // h_{-1} = 0
  if (i < 2) syncv[i] = 0u;         // barrier counter + generation
}

__global__ void k_hfinal(const float* __restrict__ hall, float* __restrict__ out) {
  int i = (int)(blockIdx.x * 256u + threadIdx.x);
  if (i < BB * HH)
    out[(size_t)BB * TT * OO + i] = hall[(size_t)TT * BB * HH + i];
}

extern "C" void kernel_launch(void* const* d_in, const int* in_sizes, int n_in,
                              void* d_out, int out_size, void* d_ws, size_t ws_size,
                              hipStream_t stream) {
  (void)in_sizes; (void)n_in; (void)out_size; (void)ws_size;
  const float* x  = (const float*)d_in[0];
  const float* W1 = (const float*)d_in[1];
  const float* b1 = (const float*)d_in[2];
  const float* W2 = (const float*)d_in[3];
  const float* b2 = (const float*)d_in[4];
  float* out = (float*)d_out;

  // Zx lives in the outs region of d_out (same size, overwritten by k_out later).
  float* zx = out;

  // Workspace layout (all chunks 32B-aligned):
  //   hall   : (T+1)*B*H f32            = 67.2 MB
  //   w1hi/lo: H*(E+H) bf16 each        = 1 MB each
  //   w2hi/lo: O*H bf16 each            = 0.5 MB each
  //   syncv  : 2 u32
  float* hall = (float*)d_ws;
  bf16_t* w1hi = (bf16_t*)(hall + (size_t)(TT + 1) * BB * HH);
  bf16_t* w1lo = w1hi + (size_t)HH * W1S;
  bf16_t* w2hi = w1lo + (size_t)HH * W1S;
  bf16_t* w2lo = w2hi + (size_t)OO * HH;
  unsigned* syncv = (unsigned*)(w2lo + (size_t)OO * HH);

  k_init<<<(BB * HH + 255) / 256, 256, 0, stream>>>(hall, syncv);
  k_pack<<<(NP1 + NP2 + 255) / 256, 256, 0, stream>>>(W1, W2, w1hi, w1lo, w2hi, w2lo);
  k_zx<<<4096, 128, 0, stream>>>(x, w1hi, w1lo, b1, zx);
  k_rnn<<<32, 128, 0, stream>>>(w1hi, w1lo, zx, hall, syncv);
  k_out<<<4096, 128, 0, stream>>>(hall, w2hi, w2lo, b2, out);
  k_hfinal<<<(BB * HH + 255) / 256, 256, 0, stream>>>(hall, out);
}
